// GGNNEncoder_78606491451793
// MI455X (gfx1250) — matrix operand order repaired and checked by reading, not compiled, not run on hardware
//
#include <hip/hip_runtime.h>
#include <hip/hip_bf16.h>

// GGNN encoder for MI455X (gfx1250, wave32).
// Dominant work: We = (relu(EA@W1+b1))@W2+b2  -> (E,1024) f16, via v_wmma_f32_16x16x32_f16.
// Then 3x message passing: per-edge 32x32 GEMV (bandwidth bound, ~312MB/step) + scatter + GRU.

typedef __attribute__((ext_vector_type(16))) _Float16 v16h;
typedef __attribute__((ext_vector_type(8)))  _Float16 v8h;
typedef __attribute__((ext_vector_type(8)))  float    v8f;

#define HD 32          // hidden dim
#define NDIM 64        // node dim
#define EDIM 16        // edge dim
#define MHID 128       // edge mlp hidden
#define NTILES 64      // 1024 / 16 output col tiles
#define KTILES 4       // 128 / 32 k tiles
#define DPAD 8         // LDS row padding (halves)

// ---------------------------------------------------------------------------
// Pack W2 (128x1024 f32, row-major) into WMMA B-fragment layout (f16).
// Fragment (nt, t): lane L holds 16 halves: b[j] = W2[t*32 + (j>>3)*16 + (L>>4)*8 + (j&7), nt*16 + (L&15)]
// Stored contiguously: W2p[((nt*4 + t)*32 + lane)*16 + j]
// ---------------------------------------------------------------------------
__global__ void pack_w2_kernel(const float* __restrict__ W2, _Float16* __restrict__ W2p) {
    int idx = blockIdx.x * blockDim.x + threadIdx.x;
    if (idx >= NTILES * KTILES * 32 * 16) return;
    int j    = idx & 15;
    int lane = (idx >> 4) & 31;
    int t    = (idx >> 9) & 3;
    int nt   = idx >> 11;
    int k    = t * 32 + ((j >> 3) << 4) + ((lane >> 4) << 3) + (j & 7);
    int n    = nt * 16 + (lane & 15);
    W2p[idx] = (_Float16)W2[k * 1024 + n];
}

// ---------------------------------------------------------------------------
// We GEMM: one block (4 waves, 128 thr) per 16-edge M-tile.
// Phase 1: U = relu(EA(16x16) @ W1(16x128) + b1) -> LDS (f16).
// Phase 2: each wave handles 16 of the 64 N-tiles; A-fragments hoisted,
//          4x v_wmma_f32_16x16x32_f16 per tile; D(+b2) staged in LDS.
// Phase 3: whole 16x1024 f16 tile streamed out with coalesced b128 stores.
// We is padded to a whole number of 16-edge tiles -> no store guards.
// ---------------------------------------------------------------------------
__global__ void we_gemm_kernel(const float* __restrict__ edge_attr,   // (E,16)
                               const float* __restrict__ W1,          // (16,128)
                               const float* __restrict__ b1,          // (128)
                               const _Float16* __restrict__ W2p,      // packed frags
                               const float* __restrict__ b2,          // (1024)
                               _Float16* __restrict__ We,             // (Epad,1024)
                               int E) {
    __shared__ _Float16 Ulds[16][MHID + DPAD];
    __shared__ _Float16 Dlds[16][1024 + DPAD];

    const int mtile = blockIdx.x;
    const int tid   = threadIdx.x;

    // ---- Phase 1: 2048 U elements / 128 threads = 16 each ----
    {
        const int m = tid >> 3;                    // 0..15 edge-in-tile
        const int e = mtile * 16 + m;
        float ea[EDIM];
        #pragma unroll
        for (int jj = 0; jj < EDIM; ++jj)
            ea[jj] = (e < E) ? edge_attr[(size_t)e * EDIM + jj] : 0.0f;
        #pragma unroll
        for (int i = 0; i < 16; ++i) {
            const int k = ((tid & 7) << 4) + i;    // 0..127
            float acc = b1[k];
            #pragma unroll
            for (int jj = 0; jj < EDIM; ++jj)
                acc += ea[jj] * W1[jj * MHID + k];
            Ulds[m][k] = (_Float16)fmaxf(acc, 0.0f);
        }
    }
    __syncthreads();

    // ---- Phase 2: WMMA ----
    const int wave = tid >> 5;
    const int lane = tid & 31;
    const int row  = lane & 15;            // A: M row
    const int koff = (lane >> 4) << 3;     // A: 0 or 8 (lane-half K offset)

    // Hoist A fragments for all 4 K-tiles (ISA 16-bit A 16x32 layout)
    v16h a[KTILES];
    #pragma unroll
    for (int t = 0; t < KTILES; ++t) {
        #pragma unroll
        for (int j = 0; j < 16; ++j)
            a[t][j] = Ulds[row][t * 32 + ((j >> 3) << 4) + koff + (j & 7)];
    }

    const int mbase   = (lane >> 4) << 3;  // D: rows 0..7 (lanes 0-15) / 8..15 (16-31)
    const int ncol_lo = lane & 15;

    for (int nt = wave; nt < NTILES; nt += 4) {
        v8f c = {};
        #pragma unroll
        for (int t = 0; t < KTILES; ++t) {
            v16h b = *(const v16h*)(W2p + ((size_t)(nt * KTILES + t) * 32 + lane) * 16);
            c = __builtin_amdgcn_wmma_f32_16x16x32_f16(
                    false, a[t], false, b, (short)0, c, false, false);
        }
        const int ncol = nt * 16 + ncol_lo;
        const float bias = b2[ncol];
        #pragma unroll
        for (int r = 0; r < 8; ++r)
            Dlds[mbase + r][ncol] = (_Float16)(c[r] + bias);
    }
    __syncthreads();

    // ---- Phase 3: coalesced b128 streaming store (16*1024 halves) ----
    // 2048 chunks of 8 halves; 128 threads x 16 chunks; consecutive tids ->
    // consecutive 16B -> full-line global_store_b128.
    _Float16* out = We + (size_t)mtile * 16 * 1024;
    #pragma unroll
    for (int i = 0; i < 16; ++i) {
        const int c    = i * 128 + tid;        // chunk id 0..2047
        const int rrow = c >> 7;               // 0..15
        const int col8 = (c & 127) << 3;       // 0..1016 step 8
        v8h v = *(const v8h*)&Dlds[rrow][col8];
        *(v8h*)(out + (size_t)rrow * 1024 + col8) = v;
    }
}

// ---------------------------------------------------------------------------
// h0 = x @ Wp + bp ; one wave per node, lane = output column
// ---------------------------------------------------------------------------
__global__ void node_proj_kernel(const float* __restrict__ x,
                                 const float* __restrict__ Wp,
                                 const float* __restrict__ bp,
                                 float* __restrict__ h, int N) {
    int g = blockIdx.x * blockDim.x + threadIdx.x;
    int node = g >> 5, lane = g & 31;
    if (node >= N) return;
    float acc = bp[lane];
    const float* xr = x + (size_t)node * NDIM;
    #pragma unroll 8
    for (int k = 0; k < NDIM; ++k)
        acc += xr[k] * Wp[k * HD + lane];
    h[(size_t)node * HD + lane] = acc;
}

__global__ void zero_kernel(float* __restrict__ p, int n) {
    int i = blockIdx.x * blockDim.x + threadIdx.x;
    if (i < n) p[i] = 0.0f;
}

// ---------------------------------------------------------------------------
// msg_e = We_e(32x32) @ h[src_e]; agg[dst_e] += msg_e. One wave per edge.
// Lane owns output row; h[src] broadcast via __shfl; We row = 64B b128 loads.
// ---------------------------------------------------------------------------
__global__ void msg_scatter_kernel(const _Float16* __restrict__ We,
                                   const float* __restrict__ h,
                                   const int* __restrict__ src,
                                   const int* __restrict__ dst,
                                   float* __restrict__ agg, int E) {
    int g = blockIdx.x * blockDim.x + threadIdx.x;
    int e = g >> 5, lane = g & 31;
    if (e >= E) return;

    const int s = src[e];
    const float hv = h[(size_t)s * HD + lane];

    const v8h* wp = (const v8h*)(We + (size_t)e * 1024 + lane * HD);
    if (e + 8 < E)   // stream-ahead hint -> global_prefetch_b8
        __builtin_prefetch((const void*)(We + (size_t)(e + 8) * 1024 + lane * HD), 0, 1);

    v8h w0 = wp[0], w1 = wp[1], w2 = wp[2], w3 = wp[3];

    float acc = 0.0f;
    #pragma unroll
    for (int j = 0; j < 8; ++j) acc += (float)w0[j] * __shfl(hv, j, 32);
    #pragma unroll
    for (int j = 0; j < 8; ++j) acc += (float)w1[j] * __shfl(hv, 8 + j, 32);
    #pragma unroll
    for (int j = 0; j < 8; ++j) acc += (float)w2[j] * __shfl(hv, 16 + j, 32);
    #pragma unroll
    for (int j = 0; j < 8; ++j) acc += (float)w3[j] * __shfl(hv, 24 + j, 32);

    atomicAdd(&agg[(size_t)dst[e] * HD + lane], acc);
}

// ---------------------------------------------------------------------------
// GRUCell (torch layout, gates [r,z,n]); one wave per node, lane = column.
// ---------------------------------------------------------------------------
__global__ void gru_kernel(const float* __restrict__ agg,
                           const float* __restrict__ h,
                           const float* __restrict__ w_ih,   // (96,32)
                           const float* __restrict__ w_hh,   // (96,32)
                           const float* __restrict__ b_ih,
                           const float* __restrict__ b_hh,
                           float* __restrict__ hout, int N) {
    int g = blockIdx.x * blockDim.x + threadIdx.x;
    int node = g >> 5, lane = g & 31;
    if (node >= N) return;

    const float a  = agg[(size_t)node * HD + lane];
    const float hv = h[(size_t)node * HD + lane];

    float gi[3], gh[3];
    #pragma unroll
    for (int gate = 0; gate < 3; ++gate) {
        const int rrow = gate * HD + lane;
        float si = b_ih[rrow], sh = b_hh[rrow];
        const float* wi = w_ih + rrow * HD;
        const float* wh = w_hh + rrow * HD;
        #pragma unroll
        for (int k = 0; k < HD; ++k) {
            si += wi[k] * __shfl(a, k, 32);
            sh += wh[k] * __shfl(hv, k, 32);
        }
        gi[gate] = si; gh[gate] = sh;
    }
    const float r  = 1.0f / (1.0f + __expf(-(gi[0] + gh[0])));
    const float z  = 1.0f / (1.0f + __expf(-(gi[1] + gh[1])));
    const float nn = tanhf(gi[2] + r * gh[2]);
    hout[(size_t)node * HD + lane] = (1.0f - z) * nn + z * hv;
}

// ---------------------------------------------------------------------------
extern "C" void kernel_launch(void* const* d_in, const int* in_sizes, int n_in,
                              void* d_out, int out_size, void* d_ws, size_t ws_size,
                              hipStream_t stream) {
    const float* x         = (const float*)d_in[0];
    const int*   edge_idx  = (const int*)d_in[1];   // (2,E): [0..E)=src, [E..2E)=dst
    const float* edge_attr = (const float*)d_in[2];
    const float* Wp        = (const float*)d_in[3];
    const float* bp        = (const float*)d_in[4];
    const float* W1        = (const float*)d_in[5];
    const float* b1        = (const float*)d_in[6];
    const float* W2        = (const float*)d_in[7];
    const float* b2        = (const float*)d_in[8];
    const float* w_ih      = (const float*)d_in[9];
    const float* w_hh      = (const float*)d_in[10];
    const float* b_ih      = (const float*)d_in[11];
    const float* b_hh      = (const float*)d_in[12];

    const int N = in_sizes[0] / NDIM;   // 10000
    const int E = in_sizes[2] / EDIM;   // 160000
    const int mtiles = (E + 15) / 16;
    const int Epad   = mtiles * 16;
    const int* src = edge_idx;
    const int* dst = edge_idx + E;

    // workspace carve-out
    char* ws = (char*)d_ws;
    size_t off = 0;
    auto carve = [&](size_t bytes) -> char* {
        off = (off + 255) & ~(size_t)255;
        char* p = ws + off; off += bytes; return p;
    };
    _Float16* W2p = (_Float16*)carve((size_t)NTILES * KTILES * 32 * 16 * sizeof(_Float16));
    _Float16* We  = (_Float16*)carve((size_t)Epad * 1024 * sizeof(_Float16));   // ~312 MB
    float*    hA  = (float*)carve((size_t)N * HD * sizeof(float));
    float*    hB  = (float*)carve((size_t)N * HD * sizeof(float));
    float*    agg = (float*)carve((size_t)N * HD * sizeof(float));

    // 1) pack W2 into WMMA B fragments
    {
        int total = NTILES * KTILES * 32 * 16;
        pack_w2_kernel<<<(total + 255) / 256, 256, 0, stream>>>(W2, W2p);
    }
    // 2) edge-network MLP -> We (f16), WMMA GEMM
    we_gemm_kernel<<<mtiles, 128, 0, stream>>>(edge_attr, W1, b1, W2p, b2, We, E);

    // 3) node projection -> hA
    node_proj_kernel<<<(N * 32 + 255) / 256, 256, 0, stream>>>(x, Wp, bp, hA, N);

    // 4) 3 message-passing steps, ping-pong h; final step writes d_out
    float* hcur = hA;
    float* hnxt = hB;
    for (int step = 0; step < 3; ++step) {
        zero_kernel<<<(N * HD + 255) / 256, 256, 0, stream>>>(agg, N * HD);
        msg_scatter_kernel<<<((size_t)E * 32 + 255) / 256, 256, 0, stream>>>(
            We, hcur, src, dst, agg, E);
        float* out = (step == 2) ? (float*)d_out : hnxt;
        gru_kernel<<<(N * 32 + 255) / 256, 256, 0, stream>>>(
            agg, hcur, w_ih, w_hh, b_ih, b_hh, out, N);
        float* t = hcur; hcur = hnxt; hnxt = t;
    }
}
